// EncoderReasoningAggregation_42537356099648
// MI455X (gfx1250) — matrix operand order repaired and checked by reading, not compiled, not run on hardware
//
#include <hip/hip_runtime.h>
#include <hip/hip_bf16.h>

// ---------------------------------------------------------------------------
// EncoderReasoningAggregation forward for MI455X (gfx1250, wave32, WMMA).
// All GEMM-shaped math runs through v_wmma_f32_16x16x32_f16 (f16 in, f32 acc).
// ---------------------------------------------------------------------------

#define NI 64
#define NC 32
#define WW 32
#define EE 1024
#define SS 256
#define BSg 512
#define LGg 16
#define RR 49
#define TT 48   // LG + W

typedef __attribute__((ext_vector_type(16))) _Float16 v16h;
typedef __attribute__((ext_vector_type(8)))  float    v8f;

__device__ __forceinline__ float sigf(float x) { return 1.f / (1.f + expf(-x)); }

__device__ __forceinline__ v8f wmma32(v16h a, v16h b, v8f c) {
  // D = A(16x32 f16) * B(32x16 f16) + C(f32)
  return __builtin_amdgcn_wmma_f32_16x16x32_f16(false, a, false, b, (short)0, c,
                                                false, false);
}

// A fragment (16x32, f16) from row-major f16 LDS tile. lane = M, packed K pairs.
__device__ __forceinline__ v16h frag_a_lds(const _Float16* A, int lda) {
  int lane = threadIdx.x & 31;
  int m = lane & 15, half = lane >> 4;
  v16h a;
#pragma unroll
  for (int j = 0; j < 8; ++j) {
    int kb = 2 * j + ((j >= 4) ? 8 : 0) + half * 8;
    a[2 * j]     = A[m * lda + kb];
    a[2 * j + 1] = A[m * lda + kb + 1];
  }
  return a;
}

// A fragment from row-major f32 global, rows >= mvalid read as zero.
__device__ __forceinline__ v16h frag_a_gl(const float* A, int lda, int mvalid) {
  int lane = threadIdx.x & 31;
  int m = lane & 15, half = lane >> 4;
  bool ok = m < mvalid;
  const float* row = A + (long)m * lda;
  v16h a;
#pragma unroll
  for (int j = 0; j < 8; ++j) {
    int kb = 2 * j + ((j >= 4) ? 8 : 0) + half * 8;
    a[2 * j]     = (_Float16)(ok ? row[kb] : 0.f);
    a[2 * j + 1] = (_Float16)(ok ? row[kb + 1] : 0.f);
  }
  return a;
}

// B fragment (32x16) from LDS stored [k][n] (K-major). lane = N.
__device__ __forceinline__ v16h frag_b_lds_kn(const _Float16* B, int ldn) {
  int lane = threadIdx.x & 31;
  int n = lane & 15, half = lane >> 4;
  v16h b;
#pragma unroll
  for (int t = 0; t < 16; ++t) b[t] = B[((t < 8 ? t : t + 8) + half * 8) * ldn + n];
  return b;
}

// B fragment from LDS stored [n][k] (row-major N x K, i.e. W^T access).
__device__ __forceinline__ v16h frag_b_lds_nt(const _Float16* B, int ldk) {
  int lane = threadIdx.x & 31;
  int n = lane & 15, half = lane >> 4;
  v16h b;
#pragma unroll
  for (int t = 0; t < 16; ++t) b[t] = B[n * ldk + (t < 8 ? t : t + 8) + half * 8];
  return b;
}

// B fragment from global f32 stored [n][k]; rows >= nvalid read as zero.
__device__ __forceinline__ v16h frag_b_gl_nt(const float* B, int ldk, int nvalid) {
  int lane = threadIdx.x & 31;
  int n = lane & 15, half = lane >> 4;
  v16h b;
#pragma unroll
  for (int t = 0; t < 16; ++t)
    b[t] = (_Float16)((n < nvalid) ? B[(long)n * ldk + (t < 8 ? t : t + 8) + half * 8] : 0.f);
  return b;
}

// ---------------------------------------------------------------------------
// Generic LDS-staged NT GEMM:  C[m,n] = act( sum_k A[m,k]*W[n,k] + bias[n] ) (+res)
// grid.x = N/128, grid.y = M/64, block = 256.
// 8 waves = 4(M) x 2(N); each wave owns a 16x64 output strip (4 accumulators):
// 4 WMMAs per wave per K-chunk between one barrier pair, 4x A-frag reuse.
// ---------------------------------------------------------------------------
__global__ __launch_bounds__(256) void gemm_nt_kernel(
    const float* __restrict__ A, int lda, const float* __restrict__ W,
    const float* __restrict__ bias, const float* __restrict__ res,
    float* __restrict__ C, int ldc, int K, int act) {
  __shared__ _Float16 As[64 * 32];
  __shared__ _Float16 Bs[32 * 128];
  int tid = threadIdx.x;
  int wave = tid >> 5, lane = tid & 31;
  int wm = wave >> 1;                 // 0..3 : M subtile
  int wn = wave & 1;                  // 0..1 : 64-wide N half
  long m0 = (long)blockIdx.y * 64;
  long n0 = (long)blockIdx.x * 128;

  // staging coordinates (A: 8 contiguous k per thread; B: 16 contiguous k)
  int am = (tid * 8) >> 5, ak = (tid * 8) & 31;
  int bn = tid >> 1, bk = (tid & 1) * 16;
  const float* arow = A + (m0 + am) * (long)lda + ak;
  const float* wrow = W + (n0 + bn) * (long)K + bk;

  v8f acc[4] = {};
  for (int kc = 0; kc < K; kc += 32) {
    { // stage A tile 64x32 (f32 -> f16)
      const float* ar = arow + kc;
#pragma unroll
      for (int j = 0; j < 8; ++j) As[tid * 8 + j] = (_Float16)ar[j];
    }
    { // stage B tile: Bs[k][n] = W[n0+n][kc+k]
      const float* wr = wrow + kc;
#pragma unroll
      for (int j = 0; j < 16; ++j) Bs[(bk + j) * 128 + bn] = (_Float16)wr[j];
    }
    if (kc + 32 < K) {  // hide HBM latency behind the WMMA chain
      __builtin_prefetch(arow + kc + 32, 0, 1);
      __builtin_prefetch(wrow + kc + 32, 0, 1);
    }
    __syncthreads();
    v16h a = frag_a_lds(As + wm * 16 * 32, 32);
#pragma unroll
    for (int s = 0; s < 4; ++s) {
      v16h b = frag_b_lds_kn(Bs + wn * 64 + s * 16, 128);
      acc[s] = wmma32(a, b, acc[s]);
    }
    __syncthreads();
  }
  int half = lane >> 4;
#pragma unroll
  for (int s = 0; s < 4; ++s) {
    long ncol = n0 + wn * 64 + s * 16 + (lane & 15);
    float bv = bias ? bias[ncol] : 0.f;
#pragma unroll
    for (int r = 0; r < 8; ++r) {
      long m = m0 + wm * 16 + r + half * 8;
      float v = acc[s][r] + bv;
      if (act == 1) v = tanhf(v);
      else if (act == 2) v = sigf(v);
      long off = m * (long)ldc + ncol;
      if (res) v += res[off];
      C[off] = v;
    }
  }
}

// ---------------------------------------------------------------------------
// RGF: Gs[b,r,s] = th[b,r,:].ph[b,s,:]   (batched A * B^T, 49x49, K=1024)
// grid (2, NI), block 256; tile t = blockIdx.x*8+wave in 4x4 grid.
// ---------------------------------------------------------------------------
__global__ __launch_bounds__(256) void rgf_gs_kernel(const float* __restrict__ th,
                                                     const float* __restrict__ ph,
                                                     float* __restrict__ Gs) {
  int b = blockIdx.y;
  int t = blockIdx.x * 8 + (threadIdx.x >> 5);
  int mt = t >> 2, nt = t & 3;
  const float* A = th + (long)b * RR * EE + mt * 16 * EE;
  const float* B = ph + (long)b * RR * EE + nt * 16 * EE;
  int mrem = RR - mt * 16, nrem = RR - nt * 16;
  v8f acc = {};
  for (int kc = 0; kc < EE; kc += 32) {
    v16h a  = frag_a_gl(A + kc, EE, mrem);
    v16h bb = frag_b_gl_nt(B + kc, EE, nrem);
    acc = wmma32(a, bb, acc);
  }
  int lane = threadIdx.x & 31;
  int n = nt * 16 + (lane & 15), half = lane >> 4;
#pragma unroll
  for (int r = 0; r < 8; ++r) {
    int m = mt * 16 + r + half * 8;
    if (m < RR && n < RR) Gs[((long)b * RR + m) * RR + n] = acc[r];
  }
}

// y[b,o,l] = sum_c w1[o,c] * Gj[b,c,l],  Gj[c,l] = c<49 ? Gs[l,c] : Gs[c-49,l]
__global__ void rgf_y_kernel(const float* __restrict__ Gs,
                             const float* __restrict__ w1, float* __restrict__ y) {
  int o = blockIdx.x, b = blockIdx.y, l = threadIdx.x;
  if (l >= RR) return;
  const float* g = Gs + (long)b * RR * RR;
  const float* wrow = w1 + o * (2 * RR);
  float s = 0.f;
  for (int c = 0; c < RR; ++c) s += wrow[c] * g[l * RR + c];
  for (int c = 0; c < RR; ++c) s += wrow[RR + c] * g[c * RR + l];
  y[((long)b * RR + o) * RR + l] = s;
}

__global__ __launch_bounds__(256) void bn_stats_kernel(const float* __restrict__ y,
                                                       float* mu, float* var) {
  int o = blockIdx.x;
  __shared__ float s1[256], s2[256];
  float a = 0.f, b2 = 0.f;
  for (int idx = threadIdx.x; idx < NI * RR; idx += 256) {
    int b = idx / RR, l = idx % RR;
    float x = y[((long)b * RR + o) * RR + l];
    a += x; b2 += x * x;
  }
  s1[threadIdx.x] = a; s2[threadIdx.x] = b2;
  __syncthreads();
  for (int s = 128; s > 0; s >>= 1) {
    if (threadIdx.x < s) { s1[threadIdx.x] += s1[threadIdx.x + s]; s2[threadIdx.x] += s2[threadIdx.x + s]; }
    __syncthreads();
  }
  if (threadIdx.x == 0) {
    float m = s1[0] / 3136.f;
    mu[o] = m;
    var[o] = s2[0] / 3136.f - m * m;
  }
}

__global__ void bn_apply_kernel(float* __restrict__ y, const float* mu,
                                const float* var, const float* g, const float* bb) {
  long idx = (long)blockIdx.x * 256 + threadIdx.x;
  if (idx >= (long)NI * RR * RR) return;
  int o = (int)((idx / RR) % RR);
  float x = (y[idx] - mu[o]) * rsqrtf(var[o] + 1e-5f) * g[o] + bb[o];
  y[idx] = tanhf(x);
}

__global__ __launch_bounds__(256) void rgf_gx_kernel(const float* __restrict__ v,
                                                     const float* __restrict__ w2w,
                                                     const float* __restrict__ w2b,
                                                     float* __restrict__ gx) {
  int row = blockIdx.x * 8 + (threadIdx.x >> 5);
  int lane = threadIdx.x & 31;
  const float* r = v + (long)row * EE;
  float s = 0.f;
  for (int k = lane; k < EE; k += 32) s += r[k] * w2w[k];
  for (int off = 16; off > 0; off >>= 1) s += __shfl_xor(s, off);
  if (lane == 0) gx[row] = tanhf(s + w2b[0]);
}

__global__ __launch_bounds__(256) void rgf_gate_kernel(float* __restrict__ v,
                                                       const float* __restrict__ y,
                                                       const float* __restrict__ gx,
                                                       const float* __restrict__ w3w,
                                                       const float* __restrict__ w3b) {
  int row = blockIdx.x;            // b*49 + r
  int b = row / RR, r = row % RR;
  float wy = w3w[0] * gx[row] + w3b[0];
  const float* yr = y + ((long)b * RR + r) * RR;
  for (int l = 0; l < RR; ++l) wy += w3w[1 + l] * yr[l];
  float gate = sigf(tanhf(wy));
  float* vr = v + (long)row * EE;
  for (int e = threadIdx.x; e < EE; e += 256) vr[e] *= gate;
}

// ---------------------------------------------------------------------------
// L2 row-normalize (row length 512)
// ---------------------------------------------------------------------------
__global__ __launch_bounds__(256) void l2norm512_kernel(const float* __restrict__ in,
                                                        float* __restrict__ out) {
  __shared__ float red[256];
  long row = blockIdx.x;
  const float* r = in + row * BSg;
  float v0 = r[threadIdx.x], v1 = r[threadIdx.x + 256];
  red[threadIdx.x] = v0 * v0 + v1 * v1;
  __syncthreads();
  for (int s = 128; s > 0; s >>= 1) {
    if (threadIdx.x < s) red[threadIdx.x] += red[threadIdx.x + s];
    __syncthreads();
  }
  float denom = sqrtf(red[0]) + 1e-8f;
  out[row * BSg + threadIdx.x]       = v0 / denom;
  out[row * BSg + threadIdx.x + 256] = v1 / denom;
}

__global__ void mask_cap_kernel(const float* __restrict__ cap, const int* __restrict__ lens,
                                float* __restrict__ capm) {
  long idx = (long)blockIdx.x * 256 + threadIdx.x;
  if (idx >= (long)NC * WW * EE) return;
  int w = (int)((idx >> 10) & 31);
  int c = (int)(idx >> 15);
  capm[idx] = (w < lens[c]) ? cap[idx] : 0.f;
}

// ---------------------------------------------------------------------------
// attn[c,i,r,w] = leaky( v[i,r,:] . capm[c,w,:] )   M=49,N=32,K=1024 per (c,i)
// grid (NI, NC), 8 waves = 4 mt x 2 nt tiles.
// ---------------------------------------------------------------------------
__global__ __launch_bounds__(256) void attn_gemm_kernel(const float* __restrict__ v,
                                                        const float* __restrict__ capm,
                                                        float* __restrict__ attn) {
  int i = blockIdx.x, c = blockIdx.y;
  int wave = threadIdx.x >> 5, lane = threadIdx.x & 31;
  int mt = wave >> 1, nt = wave & 1;
  const float* A = v + (long)i * RR * EE + mt * 16 * EE;
  const float* B = capm + (long)c * WW * EE + nt * 16 * EE;
  int mrem = RR - mt * 16;
  v8f acc = {};
  for (int kc = 0; kc < EE; kc += 32) {
    v16h a  = frag_a_gl(A + kc, EE, mrem);
    v16h bb = frag_b_gl_nt(B + kc, EE, 16);
    acc = wmma32(a, bb, acc);
  }
  int w = nt * 16 + (lane & 15), half = lane >> 4;
#pragma unroll
  for (int r = 0; r < 8; ++r) {
    int m = mt * 16 + r + half * 8;
    if (m < RR) {
      float x = acc[r];
      x = x > 0.f ? x : 0.1f * x;
      attn[(((long)c * NI + i) * RR + m) * WW + w] = x;
    }
  }
}

// per (c,i): l2 normalize over w, then softmax(12*x) over r.
__global__ __launch_bounds__(256) void attn_ns_kernel(float* __restrict__ attn) {
  long b = blockIdx.x;
  __shared__ float t[RR * WW];
  __shared__ float rn[RR];
  float* g = attn + b * RR * WW;
  for (int idx = threadIdx.x; idx < RR * WW; idx += 256) t[idx] = g[idx];
  __syncthreads();
  if (threadIdx.x < RR) {
    float s = 0.f;
    for (int w = 0; w < WW; ++w) { float x = t[threadIdx.x * WW + w]; s += x * x; }
    rn[threadIdx.x] = sqrtf(s) + 1e-8f;
  }
  __syncthreads();
  for (int idx = threadIdx.x; idx < RR * WW; idx += 256)
    t[idx] = t[idx] / rn[idx >> 5] * 12.f;
  __syncthreads();
  if (threadIdx.x < WW) {
    int w = threadIdx.x;
    float mx = -1e30f;
    for (int r = 0; r < RR; ++r) mx = fmaxf(mx, t[r * WW + w]);
    float s = 0.f;
    for (int r = 0; r < RR; ++r) { float ex = expf(t[r * WW + w] - mx); t[r * WW + w] = ex; s += ex; }
    float inv = 1.f / s;
    for (int r = 0; r < RR; ++r) t[r * WW + w] *= inv;
  }
  __syncthreads();
  for (int idx = threadIdx.x; idx < RR * WW; idx += 256) g[idx] = t[idx];
}

// ---------------------------------------------------------------------------
// sim_glob rows: d = (bembn[c,l]-ign[i,l])^2 ; sim[c,i,l,:] = d @ clip_w^T + b
// ---------------------------------------------------------------------------
__global__ __launch_bounds__(256) void simglob_kernel(const float* __restrict__ bembn,
                                                      const float* __restrict__ ign,
                                                      const float* __restrict__ clipw,
                                                      const float* __restrict__ clipb,
                                                      float* __restrict__ sim) {
  int i = blockIdx.x, c = blockIdx.y;
  __shared__ _Float16 d[LGg * BSg];
  const float* bc = bembn + (long)c * LGg * BSg;
  const float* gi = ign + (long)i * LGg * BSg;
  for (int idx = threadIdx.x; idx < LGg * BSg; idx += 256) {
    float x = bc[idx] - gi[idx];
    d[idx] = (_Float16)(x * x);
  }
  __syncthreads();
  int wave = threadIdx.x >> 5, lane = threadIdx.x & 31;
  for (int p = 0; p < 2; ++p) {
    int nt = wave + p * 8;
    v8f acc = {};
    for (int kc = 0; kc < BSg; kc += 32) {
      v16h a  = frag_a_lds(d + kc, BSg);
      v16h bb = frag_b_gl_nt(clipw + (long)nt * 16 * BSg + kc, BSg, 16);
      acc = wmma32(a, bb, acc);
    }
    int n = nt * 16 + (lane & 15), half = lane >> 4;
#pragma unroll
    for (int r = 0; r < 8; ++r) {
      int l = r + half * 8;
      sim[(((long)c * NI + i) * TT + l) * SS + n] = acc[r] + clipb[n];
    }
  }
}

// ---------------------------------------------------------------------------
// sim_rr: ctx[w,e] = sum_r attn[r,w] v[i,r,e];  d=(capm-ctx)^2;
//         sim[c,i,16+w,:] = d @ rr_w^T + rr_b      (e chunked by 512)
// ---------------------------------------------------------------------------
__global__ __launch_bounds__(256) void simrr_kernel(const float* __restrict__ v,
                                                    const float* __restrict__ capm,
                                                    const float* __restrict__ attn,
                                                    const float* __restrict__ rrw,
                                                    const float* __restrict__ rrb,
                                                    float* __restrict__ sim) {
  int i = blockIdx.x, c = blockIdx.y;
  __shared__ float at[RR * WW];
  __shared__ _Float16 d[WW * 512];
  const float* ab = attn + ((long)c * NI + i) * RR * WW;
  for (int idx = threadIdx.x; idx < RR * WW; idx += 256) at[idx] = ab[idx];
  int wave = threadIdx.x >> 5, lane = threadIdx.x & 31;
  v8f acc[2][2] = {};
  for (int chunk = 0; chunk < 2; ++chunk) {
    __syncthreads();
    const float* vb = v + (long)i * RR * EE + chunk * 512;
    const float* cb = capm + (long)c * WW * EE + chunk * 512;
    for (int idx = threadIdx.x; idx < WW * 512; idx += 256) {
      int w = idx >> 9, e = idx & 511;
      float ctx = 0.f;
      for (int r = 0; r < RR; ++r) ctx += at[r * WW + w] * vb[(long)r * EE + e];
      float x = cb[(long)w * EE + e] - ctx;
      d[idx] = (_Float16)(x * x);
    }
    __syncthreads();
    for (int kc = 0; kc < 512; kc += 32) {
#pragma unroll
      for (int ni = 0; ni < 2; ++ni) {
        int nt = wave * 2 + ni;
        v16h bb = frag_b_gl_nt(rrw + (long)nt * 16 * EE + chunk * 512 + kc, EE, 16);
#pragma unroll
        for (int mi = 0; mi < 2; ++mi) {
          v16h a = frag_a_lds(d + mi * 16 * 512 + kc, 512);
          acc[mi][ni] = wmma32(a, bb, acc[mi][ni]);
        }
      }
    }
  }
  int half = lane >> 4;
#pragma unroll
  for (int mi = 0; mi < 2; ++mi)
#pragma unroll
    for (int ni = 0; ni < 2; ++ni) {
      int n = (wave * 2 + ni) * 16 + (lane & 15);
#pragma unroll
      for (int r = 0; r < 8; ++r) {
        int w = mi * 16 + r + half * 8;
        sim[(((long)c * NI + i) * TT + LGg + w) * SS + n] = acc[mi][ni][r] + rrb[n];
      }
    }
}

// in-place l1 normalize rows of length 256
__global__ __launch_bounds__(256) void l1norm_kernel(float* __restrict__ x) {
  __shared__ float red[256];
  long row = blockIdx.x;
  float val = x[row * SS + threadIdx.x];
  red[threadIdx.x] = fabsf(val);
  __syncthreads();
  for (int s = 128; s > 0; s >>= 1) {
    if (threadIdx.x < s) red[threadIdx.x] += red[threadIdx.x + s];
    __syncthreads();
  }
  x[row * SS + threadIdx.x] = val / (red[0] + 1e-8f);
}

// ---------------------------------------------------------------------------
// GA: e[c,i,t,u] = sigmoid(q.k) * mask[c,u]    (48x48, K=256, per (c,i))
// ---------------------------------------------------------------------------
__global__ __launch_bounds__(256) void ga_e_kernel(const float* __restrict__ q,
                                                   const float* __restrict__ k,
                                                   const int* __restrict__ lens,
                                                   float* __restrict__ e) {
  long b = blockIdx.x;
  int c = (int)(b >> 6);
  const float* qb = q + b * TT * SS;
  const float* kb = k + b * TT * SS;
  __shared__ _Float16 qs[TT * SS];
  __shared__ _Float16 ks[TT * SS];
  for (int idx = threadIdx.x; idx < TT * SS; idx += 256) {
    qs[idx] = (_Float16)qb[idx];
    ks[idx] = (_Float16)kb[idx];
  }
  __syncthreads();
  int wave = threadIdx.x >> 5, lane = threadIdx.x & 31;
  int len = lens[c];
  for (int t = wave; t < 9; t += 8) {
    int mt = t / 3, nt = t % 3;
    v8f acc = {};
    for (int kc = 0; kc < SS; kc += 32) {
      v16h a  = frag_a_lds(qs + mt * 16 * SS + kc, SS);
      v16h bb = frag_b_lds_nt(ks + nt * 16 * SS + kc, SS);
      acc = wmma32(a, bb, acc);
    }
    int u = nt * 16 + (lane & 15);
    float mask = (u < LGg) ? 1.f : (((u - LGg) < len) ? 1.f : 0.f);
    int half = lane >> 4;
#pragma unroll
    for (int r = 0; r < 8; ++r) {
      int tt = mt * 16 + r + half * 8;
      e[(b * TT + tt) * TT + u] = sigf(acc[r]) * mask;
    }
  }
}

// gph = e @ s   (48x256, K=48 padded to 64)
__global__ __launch_bounds__(256) void ga_gph_kernel(const float* __restrict__ e,
                                                     const float* __restrict__ s,
                                                     float* __restrict__ gph) {
  long b = blockIdx.x;
  __shared__ _Float16 Es[TT * 64];    // [m][k] padded
  __shared__ _Float16 Sk[64 * SS];    // [k][n] padded
  const float* eb = e + b * TT * TT;
  const float* sb = s + b * TT * SS;
  for (int idx = threadIdx.x; idx < TT * 64; idx += 256) {
    int m = idx >> 6, kk = idx & 63;
    Es[idx] = (_Float16)((kk < TT) ? eb[m * TT + kk] : 0.f);
  }
  for (int idx = threadIdx.x; idx < 64 * SS; idx += 256) {
    int kk = idx >> 8, n = idx & 255;
    Sk[idx] = (_Float16)((kk < TT) ? sb[kk * SS + n] : 0.f);
  }
  __syncthreads();
  int wave = threadIdx.x >> 5, lane = threadIdx.x & 31;
  for (int mt = 0; mt < 3; ++mt) {
#pragma unroll
    for (int j = 0; j < 2; ++j) {
      int nt = wave * 2 + j;
      v8f acc = {};
      for (int kc = 0; kc < 64; kc += 32) {
        v16h a  = frag_a_lds(Es + mt * 16 * 64 + kc, 64);
        v16h bb = frag_b_lds_kn(Sk + kc * SS + nt * 16, SS);
        acc = wmma32(a, bb, acc);
      }
      int n = nt * 16 + (lane & 15), half = lane >> 4;
#pragma unroll
      for (int r = 0; r < 8; ++r) {
        int m = mt * 16 + r + half * 8;
        gph[(b * TT + m) * SS + n] = acc[r];
      }
    }
  }
}

// ---------------------------------------------------------------------------
// GRU pointwise step (h updated in place; gi/gh precomputed by gemm_nt)
// ---------------------------------------------------------------------------
__global__ void gru_step_kernel(float* __restrict__ h, const float* __restrict__ gi,
                                const float* __restrict__ gh,
                                const int* __restrict__ lens, int t) {
  long idx = (long)blockIdx.x * 256 + threadIdx.x;   // over 2048*256
  int m = (int)(idx >> 8), dd = (int)(idx & 255);
  int c = m >> 6;
  bool valid = (t < LGg) || ((t - LGg) < lens[c]);
  const float* gim = gi + (long)m * 768;
  const float* ghm = gh + (long)m * 768;
  float hv = h[idx];
  float r = sigf(gim[dd] + ghm[dd]);
  float z = sigf(gim[256 + dd] + ghm[256 + dd]);
  float n = tanhf(gim[512 + dd] + r * ghm[512 + dd]);
  float hn = (1.f - z) * n + z * hv;
  h[idx] = valid ? hn : hv;
}

__global__ void final_kernel(const float* __restrict__ h, const float* __restrict__ sw,
                             const float* __restrict__ sb, float* __restrict__ out) {
  int idx = blockIdx.x * 256 + threadIdx.x;          // 2048 = NC*NI
  int c = idx >> 6, i = idx & 63;
  const float* hr = h + (long)idx * SS;
  float s = sb[0];
  for (int d = 0; d < SS; ++d) s += hr[d] * sw[d];
  out[i * NC + c] = sigf(s);
}

// ---------------------------------------------------------------------------
// Host orchestration
// ---------------------------------------------------------------------------
extern "C" void kernel_launch(void* const* d_in, const int* in_sizes, int n_in,
                              void* d_out, int out_size, void* d_ws, size_t ws_size,
                              hipStream_t stream) {
  (void)in_sizes; (void)n_in; (void)out_size; (void)ws_size;
  const float* img_emb  = (const float*)d_in[1];
  const float* img_embg = (const float*)d_in[2];
  const float* cap_emb  = (const float*)d_in[3];
  const float* bemb     = (const float*)d_in[4];
  const int*   cap_lens = (const int*)d_in[5];
  const float* thw = (const float*)d_in[7];
  const float* thb = (const float*)d_in[8];
  const float* phw = (const float*)d_in[9];
  const float* phb = (const float*)d_in[10];
  const float* w1  = (const float*)d_in[11];
  const float* bng = (const float*)d_in[12];
  const float* bnb = (const float*)d_in[13];
  const float* w2w = (const float*)d_in[14];
  const float* w2b = (const float*)d_in[15];
  const float* w3w = (const float*)d_in[16];
  const float* w3b = (const float*)d_in[17];
  const float* qw  = (const float*)d_in[18];
  const float* qb_ = (const float*)d_in[19];
  const float* kw  = (const float*)d_in[20];
  const float* kb_ = (const float*)d_in[21];
  const float* gsw = (const float*)d_in[22];
  const float* gsb = (const float*)d_in[23];
  const float* rrw = (const float*)d_in[24];
  const float* rrb = (const float*)d_in[25];
  const float* clw = (const float*)d_in[26];
  const float* clb = (const float*)d_in[27];
  const float* smw = (const float*)d_in[28];
  const float* smb = (const float*)d_in[29];
  const float* wih = (const float*)d_in[30];
  const float* whh = (const float*)d_in[31];
  const float* bih = (const float*)d_in[32];
  const float* bhh = (const float*)d_in[33];
  float* out = (float*)d_out;
  float* ws = (float*)d_ws;

  // workspace layout (floats)
  const size_t SZ_V = (size_t)NI * RR * EE;           // 3,211,264
  float* V    = ws;
  float* TH   = V + SZ_V;
  float* PH   = TH + SZ_V;
  float* GS   = PH + SZ_V;
  float* Y    = GS + (size_t)NI * RR * RR;
  float* GX   = Y + (size_t)NI * RR * RR;
  float* MU   = GX + (size_t)NI * RR;
  float* VAR  = MU + 64;
  float* CAPM = VAR + 64;
  float* BEMBN= CAPM + (size_t)NC * WW * EE;
  float* IGN  = BEMBN + (size_t)NC * LGg * BSg;
  float* ATTN = IGN + (size_t)NI * LGg * BSg;
  float* SIM  = ATTN + (size_t)NC * NI * RR * WW;
  float* QB   = SIM + (size_t)NC * NI * TT * SS;
  float* KB   = QB + (size_t)NC * NI * TT * SS;
  float* EB   = KB + (size_t)NC * NI * TT * SS;
  float* GIB  = EB + (size_t)NC * NI * TT * TT;
  float* GHB  = GIB + (size_t)NC * NI * 768;
  float* HB   = GHB + (size_t)NC * NI * 768;

  hipMemcpyAsync(V, img_emb, SZ_V * sizeof(float), hipMemcpyDeviceToDevice, stream);

  // ---- RGF x4 ----
  for (int l = 0; l < 4; ++l) {
    gemm_nt_kernel<<<dim3(EE / 128, (NI * RR) / 64), 256, 0, stream>>>(
        V, EE, thw + (size_t)l * EE * EE, thb + l * EE, nullptr, TH, EE, EE, 1);
    gemm_nt_kernel<<<dim3(EE / 128, (NI * RR) / 64), 256, 0, stream>>>(
        V, EE, phw + (size_t)l * EE * EE, phb + l * EE, nullptr, PH, EE, EE, 1);
    rgf_gs_kernel<<<dim3(2, NI), 256, 0, stream>>>(TH, PH, GS);
    rgf_y_kernel<<<dim3(RR, NI), 64, 0, stream>>>(GS, w1 + (size_t)l * RR * 2 * RR, Y);
    bn_stats_kernel<<<RR, 256, 0, stream>>>(Y, MU, VAR);
    bn_apply_kernel<<<(NI * RR * RR + 255) / 256, 256, 0, stream>>>(
        Y, MU, VAR, bng + l * RR, bnb + l * RR);
    rgf_gx_kernel<<<(NI * RR) / 8, 256, 0, stream>>>(V, w2w + (size_t)l * EE, w2b + l, GX);
    rgf_gate_kernel<<<NI * RR, 256, 0, stream>>>(V, Y, GX, w3w + (size_t)l * (RR + 1), w3b + l);
  }

  // ---- norms + masked captions ----
  l2norm512_kernel<<<NC * LGg, 256, 0, stream>>>(bemb, BEMBN);
  l2norm512_kernel<<<NI * LGg, 256, 0, stream>>>(img_embg, IGN);
  mask_cap_kernel<<<(NC * WW * EE + 255) / 256, 256, 0, stream>>>(cap_emb, cap_lens, CAPM);

  // ---- region-word attention + sim construction ----
  attn_gemm_kernel<<<dim3(NI, NC), 256, 0, stream>>>(V, CAPM, ATTN);
  attn_ns_kernel<<<NC * NI, 256, 0, stream>>>(ATTN);
  simglob_kernel<<<dim3(NI, NC), 256, 0, stream>>>(BEMBN, IGN, clw, clb, SIM);
  simrr_kernel<<<dim3(NI, NC), 256, 0, stream>>>(V, CAPM, ATTN, rrw, rrb, SIM);
  l1norm_kernel<<<NC * NI * TT, 256, 0, stream>>>(SIM);

  // ---- GraphAggregating x3 ----
  for (int l = 0; l < 3; ++l) {
    gemm_nt_kernel<<<dim3(SS / 128, (NC * NI * TT) / 64), 256, 0, stream>>>(
        SIM, SS, qw + (size_t)l * SS * SS, qb_ + l * SS, nullptr, QB, SS, SS, 0);
    gemm_nt_kernel<<<dim3(SS / 128, (NC * NI * TT) / 64), 256, 0, stream>>>(
        SIM, SS, kw + (size_t)l * SS * SS, kb_ + l * SS, nullptr, KB, SS, SS, 0);
    ga_e_kernel<<<NC * NI, 256, 0, stream>>>(QB, KB, cap_lens, EB);
    ga_gph_kernel<<<NC * NI, 256, 0, stream>>>(EB, SIM, QB);   // gph -> QB
    gemm_nt_kernel<<<dim3(SS / 128, (NC * NI * TT) / 64), 256, 0, stream>>>(
        QB, SS, gsw + (size_t)l * SS * SS, gsb + l * SS, SIM, SIM, SS, SS, 1);
  }

  // ---- GRU over T=48 masked steps ----
  hipMemsetAsync(HB, 0, (size_t)NC * NI * SS * sizeof(float), stream);
  for (int t = 0; t < TT; ++t) {
    gemm_nt_kernel<<<dim3(768 / 128, (NC * NI) / 64), 256, 0, stream>>>(
        SIM + (size_t)t * SS, TT * SS, wih, bih, nullptr, GIB, 768, SS, 0);
    gemm_nt_kernel<<<dim3(768 / 128, (NC * NI) / 64), 256, 0, stream>>>(
        HB, SS, whh, bhh, nullptr, GHB, 768, SS, 0);
    gru_step_kernel<<<(NC * NI * SS) / 256, 256, 0, stream>>>(HB, GIB, GHB, cap_lens, t);
  }

  // ---- final similarity ----
  final_kernel<<<(NC * NI) / 256, 256, 0, stream>>>(HB, smw, smb, out);
}
